// Compressed_Attention_66039417143496
// MI455X (gfx1250) — compile-verified, matrix operand order checked by reading, not run on hardware
//
#include <hip/hip_runtime.h>

typedef __attribute__((ext_vector_type(16))) _Float16 v16h;
typedef __attribute__((ext_vector_type(8)))  float    v8f;

#define D_MODEL 1024
#define SEQ     4096
#define BATCH   4
#define HEADS   8
#define BLK     64
#define DK      128
#define M_TOTAL (BATCH * SEQ)   // 16384 rows

#define LSTR 40                 // LDS row stride in halves (32 + 8 pad, 80B: bank-conflict free)
#define ABUF (128 * LSTR)       // halves per LDS buffer (128 rows x 32 halves padded)

// ---------------------------------------------------------------------------
// Async-counter wait (CDNA5): builtin if the toolchain declares it, else asm.
// ---------------------------------------------------------------------------
#if __has_builtin(__builtin_amdgcn_s_wait_asynccnt)
#define WAIT_ASYNC(n) __builtin_amdgcn_s_wait_asynccnt(n)
#else
#define WAIT_ASYNC(n) asm volatile("s_wait_asynccnt %0" ::"i"(n) : "memory")
#endif

// Copy 64 contiguous bytes global -> LDS via 4x async b128 (ASYNCcnt += 4).
__device__ __forceinline__ void async_stage64(const _Float16* __restrict__ g,
                                              _Float16* lds_row) {
  unsigned l = (unsigned)(uintptr_t)lds_row;          // LDS byte offset (low 32 bits)
  unsigned long long a = (unsigned long long)(uintptr_t)g;
#pragma unroll
  for (int c = 0; c < 4; ++c) {
    asm volatile("global_load_async_to_lds_b128 %0, %1, off"
                 :: "v"(l + (unsigned)(c * 16)), "v"(a + (unsigned long long)(c * 16))
                 : "memory");
  }
}

// ---------------------------------------------------------------------------
// Fragment helpers (wave32, 16-bit 16x16x32 WMMA layout)
// A (16x32): lane L holds row M=L&15; half=L>>4 adds +8 to K.
//   VGPR j<4  -> K pair {2j, 2j+1} (+8*half)
//   VGPR j>=4 -> K pair {16+2(j-4), ...} (+8*half)
// B (32x16) is symmetric with N=L&15; memory addressed as p[n*ld + k].
// ---------------------------------------------------------------------------
__device__ __forceinline__ v16h load_frag16(const _Float16* __restrict__ p,
                                            int ld, int r0, int k0, int lane) {
  const int m    = lane & 15;
  const int half = lane >> 4;
  const _Float16* q = p + (size_t)(r0 + m) * ld + (k0 + 8 * half);
  v16h f;
#pragma unroll
  for (int i = 0; i < 8; ++i) f[i]     = q[i];        // VGPR 0..3: K 0..7
#pragma unroll
  for (int i = 0; i < 8; ++i) f[8 + i] = q[16 + i];   // VGPR 4..7: K 16..23
  return f;
}

__device__ __forceinline__ v8f wmma_f16(v16h a, v16h b, v8f c) {
  return __builtin_amdgcn_wmma_f32_16x16x32_f16(
      /*neg_a=*/false, a, /*neg_b=*/false, b,
      /*c_mod=*/(short)0, c, /*reuse_a=*/false, /*reuse_b=*/false);
}

// ---------------------------------------------------------------------------
// Workgroup-tiled GEMM core: 128x128 tile, 4 waves (2x2), K staged in LDS via
// double-buffered async copies. acc is this wave's 64x64 (4x4 fragment tiles).
// Xg: [M x 1024] row-major f16.  Wg: [N x 1024] row-major f16 (W transposed).
// ---------------------------------------------------------------------------
__device__ __forceinline__ void gemm_core_128(const _Float16* __restrict__ Xg,
                                              const _Float16* __restrict__ Wg,
                                              int row0, int n0, int tid,
                                              _Float16* As, _Float16* Bs,
                                              v8f acc[4][4]) {
  const int lane = tid & 31;
  const int wave = tid >> 5;
  const int wr = (wave >> 1) * 64;   // wave row offset inside 128x128 tile
  const int wc = (wave & 1) * 64;    // wave col offset inside 128x128 tile

  const _Float16* xrow = Xg + (size_t)(row0 + tid) * D_MODEL;
  const _Float16* wrow = Wg + (size_t)(n0 + tid) * D_MODEL;

  // prologue: stage k-slice 0 into buffer 0 (ASYNCcnt = 8)
  async_stage64(xrow, As + tid * LSTR);
  async_stage64(wrow, Bs + tid * LSTR);

  const int steps = D_MODEL / 32;
  for (int i = 0; i < steps; ++i) {
    const int cur = i & 1;
    if (i + 1 < steps) {
      const int nxt = cur ^ 1;
      async_stage64(xrow + (i + 1) * 32, As + nxt * ABUF + tid * LSTR);
      async_stage64(wrow + (i + 1) * 32, Bs + nxt * ABUF + tid * LSTR);
      WAIT_ASYNC(8);   // retire the 8 copies for `cur`; next 8 stay in flight
    } else {
      WAIT_ASYNC(0);
    }
    __syncthreads();   // all waves' slices for `cur` visible

    const _Float16* Ab = As + cur * ABUF;
    const _Float16* Bb = Bs + cur * ABUF;
    v16h a[4], b[4];
#pragma unroll
    for (int mt = 0; mt < 4; ++mt) a[mt] = load_frag16(Ab, LSTR, wr + 16 * mt, 0, lane);
#pragma unroll
    for (int nt = 0; nt < 4; ++nt) b[nt] = load_frag16(Bb, LSTR, wc + 16 * nt, 0, lane);
#pragma unroll
    for (int mt = 0; mt < 4; ++mt)
#pragma unroll
      for (int nt = 0; nt < 4; ++nt)
        acc[mt][nt] = wmma_f16(a[mt], b[nt], acc[mt][nt]);

    __syncthreads();   // done reading `cur` before it is refilled next iter
  }
}

// ---------------------------------------------------------------------------
// Conversion kernels
// ---------------------------------------------------------------------------
__global__ void convert_src_kernel(const float* __restrict__ in,
                                   _Float16* __restrict__ out, size_t n) {
  size_t base = ((size_t)blockIdx.x * blockDim.x + threadIdx.x) * 4;
#pragma unroll
  for (int i = 0; i < 4; ++i) {
    size_t idx = base + i;
    if (idx < n) out[idx] = (_Float16)in[idx];
  }
}

// wT[z][n][k] = W_z[k][n]  (f16, transposed so K-pairs are contiguous)
__global__ void convert_w_kernel(const float* __restrict__ Wq,
                                 const float* __restrict__ Wk,
                                 const float* __restrict__ Wv,
                                 const float* __restrict__ Wo,
                                 _Float16* __restrict__ wT) {
  int k = blockIdx.x * 16 + threadIdx.x;
  int n = blockIdx.y * 16 + threadIdx.y;
  int z = blockIdx.z;
  const float* W = (z == 0) ? Wq : (z == 1) ? Wk : (z == 2) ? Wv : Wo;
  wT[(size_t)z * D_MODEL * D_MODEL + (size_t)n * D_MODEL + k] =
      (_Float16)W[(size_t)k * D_MODEL + n];
}

// ---------------------------------------------------------------------------
// QKV projection: 128x128 workgroup tile, 4 waves. z selects Q/K/V.
// Q,K stored row-major f16 [16384 x 1024].
// V stored head-transposed: vT[(bb*HEADS+h)*DK*BLK + d*BLK + t]
// ---------------------------------------------------------------------------
__global__ void __launch_bounds__(128)
qkv_gemm_kernel(const _Float16* __restrict__ X, const _Float16* __restrict__ WT,
                const float* __restrict__ bq, const float* __restrict__ bk,
                const float* __restrict__ bv, _Float16* __restrict__ qO,
                _Float16* __restrict__ kO, _Float16* __restrict__ vT) {
  __shared__ _Float16 As[2 * ABUF];
  __shared__ _Float16 Bs[2 * ABUF];

  const int tid  = threadIdx.x;
  const int lane = tid & 31;
  const int wave = tid >> 5;
  const int row0 = blockIdx.x * 128;
  const int n0   = blockIdx.y * 128;
  const int proj = blockIdx.z;
  const _Float16* W = WT + (size_t)proj * D_MODEL * D_MODEL;

  const v8f vzero = {0.f, 0.f, 0.f, 0.f, 0.f, 0.f, 0.f, 0.f};
  v8f acc[4][4];
#pragma unroll
  for (int mt = 0; mt < 4; ++mt)
#pragma unroll
    for (int nt = 0; nt < 4; ++nt) acc[mt][nt] = vzero;

  gemm_core_128(X, W, row0, n0, tid, As, Bs, acc);

  const float* bias = (proj == 0) ? bq : (proj == 1) ? bk : bv;
  const int half = lane >> 4, nl = lane & 15;
  const int wrow = row0 + (wave >> 1) * 64;
  const int wcol = n0 + (wave & 1) * 64;
#pragma unroll
  for (int mt = 0; mt < 4; ++mt)
#pragma unroll
    for (int nt = 0; nt < 4; ++nt)
#pragma unroll
      for (int r = 0; r < 8; ++r) {
        int gm = wrow + mt * 16 + r + 8 * half;
        int gn = wcol + nt * 16 + nl;
        _Float16 hv = (_Float16)(acc[mt][nt][r] + bias[gn]);
        if (proj == 0) {
          qO[(size_t)gm * D_MODEL + gn] = hv;
        } else if (proj == 1) {
          kO[(size_t)gm * D_MODEL + gn] = hv;
        } else {
          int bbrow = gm >> 6, t = gm & 63;   // gm = b*4096 + blk*64 + t
          int hh = gn >> 7, d = gn & 127;     // gn = h*128 + d
          vT[((size_t)bbrow * HEADS + hh) * (DK * BLK) + (size_t)d * BLK + t] = hv;
        }
      }
}

// ---------------------------------------------------------------------------
// Block attention: one workgroup per (b,blk), one wave per head.
// scores(64x64) in registers -> softmax (shfl_xor within 16-lane halves)
// -> P staged in LDS -> P @ V via WMMA -> f16 attention output row-major.
// ---------------------------------------------------------------------------
__global__ void __launch_bounds__(256)
attn_kernel(const _Float16* __restrict__ Q, const _Float16* __restrict__ K,
            const _Float16* __restrict__ VT, _Float16* __restrict__ AO) {
  __shared__ _Float16 P[HEADS][BLK * BLK];   // 64 KB

  const int lane = threadIdx.x & 31;
  const int h    = threadIdx.x >> 5;         // head
  const int bb   = blockIdx.x;               // b*64 + blk
  const int row0 = bb * BLK;                 // flat token row base
  const int col0 = h * DK;
  const int half = lane >> 4, nl = lane & 15;

  const v8f vzero = {0.f, 0.f, 0.f, 0.f, 0.f, 0.f, 0.f, 0.f};
  v8f s[4][4];
#pragma unroll
  for (int mt = 0; mt < 4; ++mt)
#pragma unroll
    for (int nt = 0; nt < 4; ++nt) s[mt][nt] = vzero;

  // scores = Q_blk @ K_blk^T over dk=128
  for (int kk = 0; kk < DK; kk += 32) {
    v16h a[4], b[4];
#pragma unroll
    for (int mt = 0; mt < 4; ++mt)
      a[mt] = load_frag16(Q, D_MODEL, row0 + 16 * mt, col0 + kk, lane);
#pragma unroll
    for (int nt = 0; nt < 4; ++nt)   // B[k][n] = K[row0+n][col0+kk+k]
      b[nt] = load_frag16(K, D_MODEL, row0 + 16 * nt, col0 + kk, lane);
#pragma unroll
    for (int mt = 0; mt < 4; ++mt)
#pragma unroll
      for (int nt = 0; nt < 4; ++nt)
        s[mt][nt] = wmma_f16(a[mt], b[nt], s[mt][nt]);
  }

  // row softmax: each row M lives in 16 lanes (one half) across 4 nt tiles
  const float scale = 0.08838834764831845f;  // dk^-0.5
#pragma unroll
  for (int mt = 0; mt < 4; ++mt)
#pragma unroll
    for (int r = 0; r < 8; ++r) {
      float x0 = s[mt][0][r] * scale, x1 = s[mt][1][r] * scale;
      float x2 = s[mt][2][r] * scale, x3 = s[mt][3][r] * scale;
      float mx = fmaxf(fmaxf(x0, x1), fmaxf(x2, x3));
#pragma unroll
      for (int msk = 1; msk < 16; msk <<= 1)
        mx = fmaxf(mx, __shfl_xor(mx, msk, 32));
      float e0 = __expf(x0 - mx), e1 = __expf(x1 - mx);
      float e2 = __expf(x2 - mx), e3 = __expf(x3 - mx);
      float sum = e0 + e1 + e2 + e3;
#pragma unroll
      for (int msk = 1; msk < 16; msk <<= 1)
        sum += __shfl_xor(sum, msk, 32);
      float inv = 1.0f / sum;
      int m = mt * 16 + r + 8 * half;
      P[h][m * BLK + 0 * 16 + nl] = (_Float16)(e0 * inv);
      P[h][m * BLK + 1 * 16 + nl] = (_Float16)(e1 * inv);
      P[h][m * BLK + 2 * 16 + nl] = (_Float16)(e2 * inv);
      P[h][m * BLK + 3 * 16 + nl] = (_Float16)(e3 * inv);
    }
  __syncthreads();

  // out = P(64x64) @ V(64x128); V is [d][t] per (bb,h) so K-pairs contiguous
  const _Float16* vbase = VT + ((size_t)bb * HEADS + h) * (DK * BLK);
  for (int dt = 0; dt < 8; ++dt) {
    v8f o[4];
#pragma unroll
    for (int mt = 0; mt < 4; ++mt) o[mt] = vzero;
#pragma unroll
    for (int kk = 0; kk < BLK; kk += 32) {
      v16h b = load_frag16(vbase, BLK, dt * 16, kk, lane);  // B[k][n]=V[t=k][d=n]
#pragma unroll
      for (int mt = 0; mt < 4; ++mt) {
        v16h a = load_frag16(&P[h][0], BLK, mt * 16, kk, lane);
        o[mt] = wmma_f16(a, b, o[mt]);
      }
    }
#pragma unroll
    for (int mt = 0; mt < 4; ++mt)
#pragma unroll
      for (int r = 0; r < 8; ++r) {
        int gm = row0 + mt * 16 + r + 8 * half;
        int gn = col0 + dt * 16 + nl;
        AO[(size_t)gm * D_MODEL + gn] = (_Float16)o[mt][r];
      }
  }
}

// ---------------------------------------------------------------------------
// Output projection: attn(f16) @ Wo^T(f16) + bo -> fp32 d_out
// ---------------------------------------------------------------------------
__global__ void __launch_bounds__(128)
out_gemm_kernel(const _Float16* __restrict__ X, const _Float16* __restrict__ WoT,
                const float* __restrict__ bo, float* __restrict__ out) {
  __shared__ _Float16 As[2 * ABUF];
  __shared__ _Float16 Bs[2 * ABUF];

  const int tid  = threadIdx.x;
  const int lane = tid & 31;
  const int wave = tid >> 5;
  const int row0 = blockIdx.x * 128;
  const int n0   = blockIdx.y * 128;

  const v8f vzero = {0.f, 0.f, 0.f, 0.f, 0.f, 0.f, 0.f, 0.f};
  v8f acc[4][4];
#pragma unroll
  for (int mt = 0; mt < 4; ++mt)
#pragma unroll
    for (int nt = 0; nt < 4; ++nt) acc[mt][nt] = vzero;

  gemm_core_128(X, WoT, row0, n0, tid, As, Bs, acc);

  const int half = lane >> 4, nl = lane & 15;
  const int wrow = row0 + (wave >> 1) * 64;
  const int wcol = n0 + (wave & 1) * 64;
#pragma unroll
  for (int mt = 0; mt < 4; ++mt)
#pragma unroll
    for (int nt = 0; nt < 4; ++nt)
#pragma unroll
      for (int r = 0; r < 8; ++r) {
        int gm = wrow + mt * 16 + r + 8 * half;
        int gn = wcol + nt * 16 + nl;
        out[(size_t)gm * D_MODEL + gn] = acc[mt][nt][r] + bo[gn];
      }
}

// ---------------------------------------------------------------------------
extern "C" void kernel_launch(void* const* d_in, const int* in_sizes, int n_in,
                              void* d_out, int out_size, void* d_ws, size_t ws_size,
                              hipStream_t stream) {
  const float* src = (const float*)d_in[0];
  const float* Wq  = (const float*)d_in[1];
  const float* bq  = (const float*)d_in[2];
  const float* Wk  = (const float*)d_in[3];
  const float* bk  = (const float*)d_in[4];
  const float* Wv  = (const float*)d_in[5];
  const float* bv  = (const float*)d_in[6];
  const float* Wo  = (const float*)d_in[7];
  const float* bo  = (const float*)d_in[8];
  float* out = (float*)d_out;

  const size_t actN = (size_t)M_TOTAL * D_MODEL;   // 16,777,216 halves
  const size_t wN   = (size_t)4 * D_MODEL * D_MODEL;

  _Float16* srcH = (_Float16*)d_ws;        // also reused as attention output
  _Float16* wT   = srcH + actN;            // 4 transposed f16 weight matrices
  _Float16* qH   = wT + wN;
  _Float16* kH   = qH + actN;
  _Float16* vT   = kH + actN;              // head-transposed V
  _Float16* attn = srcH;                   // alias: src dead after QKV GEMM

  // 1. fp32 -> f16 conversions
  convert_src_kernel<<<(unsigned)(actN / (256 * 4)), 256, 0, stream>>>(src, srcH, actN);
  convert_w_kernel<<<dim3(D_MODEL / 16, D_MODEL / 16, 4), dim3(16, 16), 0, stream>>>(
      Wq, Wk, Wv, Wo, wT);

  // 2. fused QKV projections (128x128 tile per workgroup, async LDS staging)
  qkv_gemm_kernel<<<dim3(M_TOTAL / 128, D_MODEL / 128, 3), 128, 0, stream>>>(
      srcH, wT, bq, bk, bv, qH, kH, vT);

  // 3. block-local attention (one workgroup per (b,blk), one wave per head)
  attn_kernel<<<dim3(BATCH * (SEQ / BLK)), 256, 0, stream>>>(qH, kH, vT, attn);

  // 4. output projection -> fp32
  out_gemm_kernel<<<dim3(M_TOTAL / 128, D_MODEL / 128), 128, 0, stream>>>(
      attn, wT + (size_t)3 * D_MODEL * D_MODEL, bo, out);
}
